// TSGNN_27152783245778
// MI455X (gfx1250) — matrix operand order repaired and checked by reading, not compiled, not run on hardware
//
#include <hip/hip_runtime.h>
#include <hip/hip_bf16.h>
#include <math.h>

typedef __attribute__((ext_vector_type(16))) __bf16 v16bf;
typedef __attribute__((ext_vector_type(8)))  float  v8f;

union FragU { uint4 q[2]; v16bf v; };

__device__ __forceinline__ unsigned pack_bf16(float a, float b) {
  union { __bf16 h[2]; unsigned u; } x;
  x.h[0] = (__bf16)a; x.h[1] = (__bf16)b;
  return x.u;
}

__device__ __forceinline__ float sigmoidf_(float x) { return 1.0f / (1.0f + __expf(-x)); }

__device__ __forceinline__ float wred_sum(float v) {
  #pragma unroll
  for (int m = 16; m > 0; m >>= 1) v += __shfl_xor(v, m, 32);
  return v;
}
__device__ __forceinline__ float wred_max(float v) {
  #pragma unroll
  for (int m = 16; m > 0; m >>= 1) v = fmaxf(v, __shfl_xor(v, m, 32));
  return v;
}

#define NSTK 2048
#define HDIM 128
#define TSTEPS 50

// ---------------------------------------------------------------------------
// Pre-swizzle W_hh^T / W_ih^T (fp32) into bf16 WMMA B-fragment layout.
// B 16-bit 32x16 layout: lanes 0-15 hold col n=lane, K=0..7,16..23 (2/VGPR);
// lanes 16-31 hold K=8..15,24..31.  Fragment fi stored as 256 dwords: lane*8 + j.
// ---------------------------------------------------------------------------
__global__ __launch_bounds__(256) void prep_weights_k(
    const float* __restrict__ Wih, const float* __restrict__ Whh,
    unsigned* __restrict__ fragHH, unsigned* __restrict__ fragIH)
{
  int tid = blockIdx.x * 256 + threadIdx.x;
  if (tid < 32768) {                       // W_hh: 32 ntiles x 4 kchunks
    int j = tid & 7, lane = (tid >> 3) & 31, kc = (tid >> 8) & 3, nt = tid >> 10;
    int n = nt * 16 + (lane & 15);
    int K = kc * 32 + ((j < 4) ? ((lane >> 4) * 8 + 2 * j) : (16 + (lane >> 4) * 8 + 2 * (j - 4)));
    fragHH[tid] = pack_bf16(Whh[n * HDIM + K], Whh[n * HDIM + K + 1]);
  } else if (tid < 40960) {                // W_ih: 32 ntiles x 1 kchunk (K padded to 32)
    int t2 = tid - 32768;
    int j = t2 & 7, lane = (t2 >> 3) & 31, nt = t2 >> 8;
    int n = nt * 16 + (lane & 15);
    int K = (j < 4) ? ((lane >> 4) * 8 + 2 * j) : (16 + (lane >> 4) * 8 + 2 * (j - 4));
    unsigned val = 0u;
    if (K < 16) val = pack_bf16(Wih[n * 16 + K], Wih[n * 16 + K + 1]);
    fragIH[t2] = val;
  }
}

// ---------------------------------------------------------------------------
// LSTM: 128 blocks x 16 stocks.  8 waves; wave w owns h-cols 16w..16w+15 of
// all 4 gates -> gate math fully in registers.  W in LDS (bf16 fragments),
// staged via GLOBAL_LOAD_ASYNC_TO_LDS_B128 (ASYNCcnt path, bypasses VGPRs).
// ---------------------------------------------------------------------------
__global__ __launch_bounds__(256) void lstm_k(
    const float* __restrict__ x, const unsigned* __restrict__ fragHH,
    const unsigned* __restrict__ fragIH,
    const float* __restrict__ b_ih, const float* __restrict__ b_hh,
    float* __restrict__ se)
{
  extern __shared__ unsigned smem[];
  unsigned* s_Whh = smem;            // 32768 dwords (128 KB) @ LDS offset 0
  unsigned* s_Wih = smem + 32768;    //  8192 dwords ( 32 KB) @ LDS offset 131072
  unsigned* s_h   = smem + 40960;    //  1024 dwords (16 rows x 64 pair-dwords)
  unsigned* s_x   = smem + 41984;    //   128 dwords (16 rows x 8 pair-dwords)

  const int tid  = threadIdx.x;
  const int lane = tid & 31;
  const int wv   = tid >> 5;
  const int blk  = blockIdx.x;

  // stage 160KB of weight fragments (HH 128KB + IH 32KB, contiguous in both
  // global workspace and LDS) directly into LDS: async copy, no VGPR bounce.
  {
    const unsigned char* g = (const unsigned char*)fragHH;
    for (int i = tid; i < 10240; i += 256) {
      unsigned lds_off = (unsigned)i * 16u;                       // dsaddr (LDS-base relative)
      unsigned long long ga = (unsigned long long)(g + (size_t)i * 16u);
      asm volatile("global_load_async_to_lds_b128 %0, %1, off"
                   :: "v"(lds_off), "v"(ga) : "memory");
    }
  }
  for (int i = tid; i < 1024; i += 256) s_h[i] = 0u;   // h0 = 0

  const int col = lane & 15;
  const int hi  = lane >> 4;
  const int jh  = wv * 16 + col;          // h-column owned by this lane
  float bias[4];
  #pragma unroll
  for (int g = 0; g < 4; ++g) bias[g] = b_ih[g * HDIM + jh] + b_hh[g * HDIM + jh];

  float c[8], h[8];
  #pragma unroll
  for (int v = 0; v < 8; ++v) { c[v] = 0.f; h[v] = 0.f; }

  asm volatile("s_wait_asynccnt 0x0" ::: "memory");    // weight tiles landed in LDS
  __syncthreads();

  for (int t = 0; t < TSTEPS; ++t) {
    // load x_t tile as packed bf16 pairs
    if (tid < 128) {
      int m = tid >> 3, p = tid & 7;
      size_t base = ((size_t)t * NSTK + (size_t)(blk * 16 + m)) * 16 + 2 * p;
      s_x[m * 8 + p] = pack_bf16(x[base], x[base + 1]);
    }
    __syncthreads();   // x + previous h visible

    // A fragments (16-bit 16x32 layout: lane m = lane&15, K-half by lane>>4)
    FragU ax, ah[4];
    ax.q[0] = *(const uint4*)&s_x[(lane & 15) * 8 + hi * 4];
    ax.q[1] = make_uint4(0u, 0u, 0u, 0u);          // K=16..31 zero pad
    #pragma unroll
    for (int kc = 0; kc < 4; ++kc) {
      const unsigned* row = s_h + (lane & 15) * 64 + kc * 16 + hi * 4;
      ah[kc].q[0] = *(const uint4*)row;
      ah[kc].q[1] = *(const uint4*)(row + 8);
    }

    v8f z[4];
    #pragma unroll
    for (int g = 0; g < 4; ++g) {
      const int nt = g * 8 + wv;
      v8f acc;
      #pragma unroll
      for (int v = 0; v < 8; ++v) acc[v] = bias[g];
      FragU b;
      b.q[0] = *(const uint4*)&s_Wih[nt * 256 + lane * 8];
      b.q[1] = *(const uint4*)&s_Wih[nt * 256 + lane * 8 + 4];
      acc = __builtin_amdgcn_wmma_f32_16x16x32_bf16(false, ax.v, false, b.v,
                                                    (short)0, acc, false, false);
      #pragma unroll
      for (int kc = 0; kc < 4; ++kc) {
        FragU bh;
        bh.q[0] = *(const uint4*)&s_Whh[(nt * 4 + kc) * 256 + lane * 8];
        bh.q[1] = *(const uint4*)&s_Whh[(nt * 4 + kc) * 256 + lane * 8 + 4];
        acc = __builtin_amdgcn_wmma_f32_16x16x32_bf16(false, ah[kc].v, false, bh.v,
                                                      (short)0, acc, false, false);
      }
      z[g] = acc;
    }

    // gate math in registers (C/D layout: vgpr v -> row m = v + 8*hi, col jh)
    #pragma unroll
    for (int v = 0; v < 8; ++v) {
      float zi = z[0][v], zf = z[1][v], zg = z[2][v], zo = z[3][v];
      float cv = sigmoidf_(zf) * c[v] + sigmoidf_(zi) * tanhf(zg);
      c[v] = cv;
      h[v] = sigmoidf_(zo) * tanhf(cv);
    }
    __syncthreads();   // all A-fragment reads of s_h / s_x done

    __bf16* hb = (__bf16*)s_h;
    #pragma unroll
    for (int v = 0; v < 8; ++v) {
      int m = v + hi * 8;
      hb[m * HDIM + jh] = (__bf16)h[v];
    }
    // next-iteration barrier makes these writes visible before reads
  }

  #pragma unroll
  for (int v = 0; v < 8; ++v) {
    int m = v + hi * 8;
    se[(size_t)(blk * 16 + m) * HDIM + jh] = h[v];
  }
}

// ---------------------------------------------------------------------------
// Neighbor attention: one wave per (r,n); lane = neighbor k.
// ---------------------------------------------------------------------------
__global__ __launch_bounds__(256) void attn_k(
    const float* __restrict__ se, const int* __restrict__ nb,
    const float* __restrict__ rel_num, const float* __restrict__ w_att,
    const float* __restrict__ b_att, float* __restrict__ rel_rep)
{
  int wid  = blockIdx.x * 8 + (threadIdx.x >> 5);
  int lane = threadIdx.x & 31;
  int r = wid >> 11;
  int n = wid & (NSTK - 1);

  float sd = 0.f;                                   // se[n] . w_self
  #pragma unroll
  for (int g = 0; g < 4; ++g) {
    int hh = g * 32 + lane;
    sd += se[(size_t)n * HDIM + hh] * w_att[HDIM + hh];
  }
  sd = wred_sum(sd);

  int idx = nb[((size_t)r * NSTK + n) * 32 + lane];
  float sc = 0.f;
  if (idx != 0) {
    const float* e = se + (size_t)(idx - 1) * HDIM;
    #pragma unroll 8
    for (int hh = 0; hh < HDIM; ++hh) sc += e[hh] * w_att[hh];
  }
  sc += sd + w_att[2 * HDIM + r] + b_att[0];

  float mx  = wred_max(sc);
  float ev  = __expf(sc - mx);
  float sm  = wred_sum(ev);
  float wgt = ev / sm;

  float inv = 1.0f / (rel_num[(size_t)r * NSTK + n] + 1e-10f);
  #pragma unroll
  for (int g = 0; g < 4; ++g) {
    float acc = 0.f;
    for (int k = 0; k < 32; ++k) {
      float wk = __shfl(wgt, k, 32);
      int   ik = __shfl(idx, k, 32);
      if (ik != 0) acc += wk * se[(size_t)(ik - 1) * HDIM + g * 32 + lane];
    }
    rel_rep[((size_t)r * NSTK + n) * HDIM + g * 32 + lane] = acc * inv;
  }
}

// ---------------------------------------------------------------------------
// Relation softmax + mean + residual + 128->1 projection: one wave per n.
// ---------------------------------------------------------------------------
__global__ __launch_bounds__(256) void final_k(
    const float* __restrict__ se, const float* __restrict__ rel_rep,
    const float* __restrict__ w_rel, const float* __restrict__ b_rel,
    const float* __restrict__ w_fc1, const float* __restrict__ b_fc1,
    float* __restrict__ out)
{
  int n    = blockIdx.x * 8 + (threadIdx.x >> 5);
  int lane = threadIdx.x & 31;

  float ser[4], rr[8][4], rs[8];
  #pragma unroll
  for (int g = 0; g < 4; ++g) ser[g] = se[(size_t)n * HDIM + g * 32 + lane];

  #pragma unroll
  for (int r = 0; r < 8; ++r) {
    float p = 0.f;
    #pragma unroll
    for (int g = 0; g < 4; ++g) {
      rr[r][g] = rel_rep[((size_t)r * NSTK + n) * HDIM + g * 32 + lane];
      p += rr[r][g] * w_rel[g * 32 + lane];
    }
    p = wred_sum(p);
    rs[r] = p + w_rel[HDIM + r] + b_rel[0];
  }
  float mx = rs[0];
  #pragma unroll
  for (int r = 1; r < 8; ++r) mx = fmaxf(mx, rs[r]);
  float s = 0.f;
  #pragma unroll
  for (int r = 0; r < 8; ++r) { rs[r] = __expf(rs[r] - mx); s += rs[r]; }
  float inv = 1.0f / (s * 8.0f);            // softmax normalize + mean over R

  float pred = 0.f;
  #pragma unroll
  for (int g = 0; g < 4; ++g) {
    float u = 0.f;
    #pragma unroll
    for (int r = 0; r < 8; ++r) u += rs[r] * rr[r][g];
    u = u * inv + ser[g];
    pred += u * w_fc1[g * 32 + lane];
  }
  pred = wred_sum(pred);
  if (lane == 0) out[n] = pred + b_fc1[0];
}

// ---------------------------------------------------------------------------
extern "C" void kernel_launch(void* const* d_in, const int* in_sizes, int n_in,
                              void* d_out, int out_size, void* d_ws, size_t ws_size,
                              hipStream_t stream)
{
  const float* stock_hist = (const float*)d_in[0];
  const int*   neighbors  = (const int*)d_in[1];
  const float* rel_num    = (const float*)d_in[2];
  const float* W_ih       = (const float*)d_in[3];
  const float* W_hh       = (const float*)d_in[4];
  const float* b_ih       = (const float*)d_in[5];
  const float* b_hh       = (const float*)d_in[6];
  const float* w_att      = (const float*)d_in[7];
  const float* b_att      = (const float*)d_in[8];
  const float* w_rel      = (const float*)d_in[9];
  const float* b_rel      = (const float*)d_in[10];
  const float* w_fc1      = (const float*)d_in[11];
  const float* b_fc1      = (const float*)d_in[12];

  unsigned* fragHH = (unsigned*)d_ws;          // 32768 dwords
  unsigned* fragIH = fragHH + 32768;           //  8192 dwords (contiguous after HH)
  float* se        = (float*)(fragIH + 8192);  // 2048*128 floats
  float* rel_rep   = se + (size_t)NSTK * HDIM; // 8*2048*128 floats

  prep_weights_k<<<160, 256, 0, stream>>>(W_ih, W_hh, fragHH, fragIH);

  size_t shmem = (size_t)42112 * 4;            // ~165 KB dynamic LDS (CDNA5: 320 KB/WGP)
  lstm_k<<<NSTK / 16, 256, shmem, stream>>>(stock_hist, fragHH, fragIH, b_ih, b_hh, se);

  attn_k<<<(8 * NSTK) / 8, 256, 0, stream>>>(se, neighbors, rel_num, w_att, b_att, rel_rep);

  final_k<<<NSTK / 8, 256, 0, stream>>>(se, rel_rep, w_rel, b_rel, w_fc1, b_fc1,
                                        (float*)d_out);
}
